// TransformerLayer_49185965474492
// MI455X (gfx1250) — compile-verified, matrix operand order checked by reading, not run on hardware
//
#include <hip/hip_runtime.h>
#include <hip/hip_bf16.h>
#include <cmath>

// ---------------------------------------------------------------------------
// Transformer layer (B=1, S=2048, D=2048, H=16, HD=128, HID=8192), fp32 I/O.
// All GEMMs run on v_wmma_f32_16x16x32_bf16 (bf16 in, f32 accumulate).
// GEMM K-loop is double-buffered; global->LDS uses the CDNA5 async path
// (GLOBAL_LOAD_ASYNC_TO_LDS_B128 + s_wait_asynccnt) when the builtin exists.
// Attention is flash-style (no SxS score tensor in HBM).
// ---------------------------------------------------------------------------

#define SEQ 2048
#define DIM 2048
#define NHEADS 16
#define HDIM 128
#define HID 8192

typedef unsigned short u16;
typedef __attribute__((ext_vector_type(16))) __bf16 v16bf;
typedef __attribute__((ext_vector_type(8)))  float  v8f;
typedef __attribute__((ext_vector_type(4)))  int    v4i;

union FragBF {
    v16bf v;
    uint4 q[2];
};

// ---- CDNA5 async global->LDS path (compile-guarded) -----------------------
#if defined(__gfx1250__) && __has_builtin(__builtin_amdgcn_global_load_async_to_lds_b128)
#define USE_ASYNC 1
#else
#define USE_ASYNC 0
#endif

#if USE_ASYNC
#if __has_builtin(__builtin_amdgcn_s_wait_asynccnt)
#define ASYNC_WAIT(n) __builtin_amdgcn_s_wait_asynccnt(n)
#else
#define ASYNC_WAIT(n) asm volatile("s_wait_asynccnt %0" ::"i"(n) : "memory")
#endif
typedef __attribute__((address_space(1))) v4i gv4i;
typedef __attribute__((address_space(3))) v4i lv4i;
#else
#define ASYNC_WAIT(n) ((void)0)
#endif

// copy 16 bytes global -> LDS (async when available)
__device__ __forceinline__ void cp16(const u16* g, u16* l) {
#if USE_ASYNC
    __builtin_amdgcn_global_load_async_to_lds_b128((gv4i*)g, (lv4i*)l, 0, 0);
#else
    *(uint4*)l = *(const uint4*)g;
#endif
}

__device__ __forceinline__ u16 f2bf(float f) {
    unsigned int u = __float_as_uint(f);
    u += 0x7fffu + ((u >> 16) & 1u);       // round-to-nearest-even
    return (u16)(u >> 16);
}

// ---------------------------------------------------------------------------
// Weight transpose + fp32 -> bf16 convert:  W[K][N] -> Wt[N][K]
// ---------------------------------------------------------------------------
__global__ __launch_bounds__(256) void transpose_cvt_kernel(
    const float* __restrict__ W, u16* __restrict__ Wt, int K, int N) {
    __shared__ float tile[32][33];
    const int n0 = blockIdx.x * 32;
    const int k0 = blockIdx.y * 32;
    const int tx = threadIdx.x;           // 0..31
    const int ty = threadIdx.y;           // 0..7
#pragma unroll
    for (int i = 0; i < 32; i += 8)
        tile[ty + i][tx] = W[(size_t)(k0 + ty + i) * N + n0 + tx];
    __syncthreads();
#pragma unroll
    for (int i = 0; i < 32; i += 8)
        Wt[(size_t)(n0 + ty + i) * K + k0 + tx] = f2bf(tile[tx][ty + i]);
}

// ---------------------------------------------------------------------------
// RMSNorm row kernel: out_bf16[row][:] = x/rms(x) * w
// ---------------------------------------------------------------------------
__global__ __launch_bounds__(256) void rmsnorm_kernel(
    const float* __restrict__ x, const float* __restrict__ w,
    u16* __restrict__ out) {
    const int row = blockIdx.x;
    const float* xr = x + (size_t)row * DIM;
    float ss = 0.f;
    for (int i = threadIdx.x; i < DIM; i += 256) {
        float v = xr[i];
        ss += v * v;
    }
#pragma unroll
    for (int m = 16; m >= 1; m >>= 1) ss += __shfl_xor(ss, m, 32);
    __shared__ float red[8];
    if ((threadIdx.x & 31) == 0) red[threadIdx.x >> 5] = ss;
    __syncthreads();
    float tot = 0.f;
#pragma unroll
    for (int i = 0; i < 8; ++i) tot += red[i];
    const float inv = rsqrtf(tot * (1.0f / DIM) + 1e-6f);
    for (int i = threadIdx.x; i < DIM; i += 256)
        out[(size_t)row * DIM + i] = f2bf(xr[i] * inv * w[i]);
}

// ---------------------------------------------------------------------------
// RoPE + fp32->bf16 (scale folds in 1/sqrt(HDIM) for Q). One thread per pair.
// ---------------------------------------------------------------------------
__global__ __launch_bounds__(256) void rope_cvt_kernel(
    const float* __restrict__ src, const float* __restrict__ cosf,
    const float* __restrict__ sinf, u16* __restrict__ dst, float scale) {
    const int gid = blockIdx.x * 256 + threadIdx.x;   // over SEQ * DIM/2
    const int s = gid >> 10;                          // DIM/2 = 1024 pairs/row
    const int p = gid & 1023;
    const int dpair = p & 63;                         // HDIM/2 = 64
    const int col = ((p >> 6) << 7) + (dpair << 1);   // head*128 + 2*dpair
    const size_t base = (size_t)s * DIM + col;
    const float r = src[base], im = src[base + 1];
    const float c = cosf[s * 64 + dpair];
    const float sn = sinf[s * 64 + dpair];
    dst[base]     = f2bf((r * c - im * sn) * scale);
    dst[base + 1] = f2bf((r * sn + im * c) * scale);
}

// ---------------------------------------------------------------------------
// V transpose per head: vt[head*128+d][s] = bf16(v[s][head*128+d])
// ---------------------------------------------------------------------------
__global__ __launch_bounds__(256) void vtrans_kernel(
    const float* __restrict__ vf, u16* __restrict__ vt) {
    const int gid = blockIdx.x * 256 + threadIdx.x;   // over SEQ*DIM
    const int dg = gid >> 11;                         // 0..2047 (head*128+d)
    const int s = gid & 2047;
    vt[gid] = f2bf(vf[(size_t)s * DIM + dg]);
}

// ---------------------------------------------------------------------------
// bf16 GEMM:  C[M][N] (f32, optional +resid) = A[M][K](bf16) @ Bt[N][K]^T
// Block tile 128x128x32, 8 waves (2x4), wave tile 64x32 = 4x2 WMMA frags.
// Double-buffered LDS, async global->LDS copies.
// ---------------------------------------------------------------------------
#define BM 128
#define BN 128
#define BK 32
#define LDSK (BK + 8)   // pad: conflict-free 16-row x 16B fragment reads

__device__ __forceinline__ v8f wmma_bf16(const FragBF& a, const FragBF& b, v8f c) {
    return __builtin_amdgcn_wmma_f32_16x16x32_bf16(
        false, a.v, false, b.v, (short)0, c, false, false);
}

template <bool RESID>
__global__ __launch_bounds__(256) void gemm_bf16_f32(
    const u16* __restrict__ A, const u16* __restrict__ Bt,
    const float* __restrict__ resid, float* __restrict__ C,
    int M, int N, int K) {
    __shared__ __align__(16) u16 As[2][BM][LDSK];
    __shared__ __align__(16) u16 Bs[2][BN][LDSK];
    const int tid = threadIdx.x;
    const int bm = blockIdx.y * BM;
    const int bn = blockIdx.x * BN;
    const int wave = tid >> 5, lane = tid & 31;
    const int wm = (wave >> 2) << 6;   // 0 / 64
    const int wn = (wave & 3) << 5;    // 0 / 32 / 64 / 96
    const int lm = lane & 15, lh = lane >> 4;

    // per-thread copy assignment: 512 16B-chunks per tile / 256 threads = 2
    auto copy_tile = [&](int kt, int buf) {
#pragma unroll
        for (int p = 0; p < 2; ++p) {
            const int id = (p << 8) + tid;
            const int r = id >> 2;
            const int c = (id & 3) << 3;
            cp16(A + (size_t)(bm + r) * K + (size_t)kt * BK + c, &As[buf][r][c]);
            cp16(Bt + (size_t)(bn + r) * K + (size_t)kt * BK + c, &Bs[buf][r][c]);
        }
    };

    const int nk = K / BK;
    v8f acc[4][2] = {};
    copy_tile(0, 0);
    for (int kt = 0; kt < nk; ++kt) {
        const int cur = kt & 1;
        if (kt + 1 < nk) {
            copy_tile(kt + 1, cur ^ 1);
            ASYNC_WAIT(4);          // own 4 copies for tile kt have landed
        } else {
            ASYNC_WAIT(0);
        }
        __syncthreads();

        FragBF a[4], b[2];
#pragma unroll
        for (int i = 0; i < 4; ++i) {
            const int row = wm + (i << 4) + lm;
            a[i].q[0] = *(const uint4*)&As[cur][row][lh << 3];
            a[i].q[1] = *(const uint4*)&As[cur][row][16 + (lh << 3)];
        }
#pragma unroll
        for (int j = 0; j < 2; ++j) {
            const int col = wn + (j << 4) + lm;
            b[j].q[0] = *(const uint4*)&Bs[cur][col][lh << 3];
            b[j].q[1] = *(const uint4*)&Bs[cur][col][16 + (lh << 3)];
        }
#pragma unroll
        for (int i = 0; i < 4; ++i)
#pragma unroll
            for (int j = 0; j < 2; ++j)
                acc[i][j] = wmma_bf16(a[i], b[j], acc[i][j]);
        __syncthreads();
    }
#pragma unroll
    for (int i = 0; i < 4; ++i)
#pragma unroll
        for (int j = 0; j < 2; ++j)
#pragma unroll
            for (int r = 0; r < 8; ++r) {
                const int row = bm + wm + (i << 4) + (lh << 3) + r;
                const int col = bn + wn + (j << 4) + lm;
                float v = acc[i][j][r];
                if (RESID) v += resid[(size_t)row * N + col];
                C[(size_t)row * N + col] = v;
            }
}

// ---------------------------------------------------------------------------
// Fused SwiGLU dual GEMM:  G = bf16( silu(A@W1t^T) * (A@W3t^T) )
// ---------------------------------------------------------------------------
__global__ __launch_bounds__(256) void gemm_swiglu(
    const u16* __restrict__ A, const u16* __restrict__ B1t,
    const u16* __restrict__ B3t, u16* __restrict__ G,
    int M, int N, int K) {
    __shared__ __align__(16) u16 As[2][BM][LDSK];
    __shared__ __align__(16) u16 B1s[2][BN][LDSK];
    __shared__ __align__(16) u16 B3s[2][BN][LDSK];
    const int tid = threadIdx.x;
    const int bm = blockIdx.y * BM;
    const int bn = blockIdx.x * BN;
    const int wave = tid >> 5, lane = tid & 31;
    const int wm = (wave >> 2) << 6;
    const int wn = (wave & 3) << 5;
    const int lm = lane & 15, lh = lane >> 4;

    auto copy_tile = [&](int kt, int buf) {
#pragma unroll
        for (int p = 0; p < 2; ++p) {
            const int id = (p << 8) + tid;
            const int r = id >> 2;
            const int c = (id & 3) << 3;
            cp16(A   + (size_t)(bm + r) * K + (size_t)kt * BK + c, &As[buf][r][c]);
            cp16(B1t + (size_t)(bn + r) * K + (size_t)kt * BK + c, &B1s[buf][r][c]);
            cp16(B3t + (size_t)(bn + r) * K + (size_t)kt * BK + c, &B3s[buf][r][c]);
        }
    };

    const int nk = K / BK;
    v8f acc1[4][2] = {}, acc3[4][2] = {};
    copy_tile(0, 0);
    for (int kt = 0; kt < nk; ++kt) {
        const int cur = kt & 1;
        if (kt + 1 < nk) {
            copy_tile(kt + 1, cur ^ 1);
            ASYNC_WAIT(6);
        } else {
            ASYNC_WAIT(0);
        }
        __syncthreads();

        FragBF a[4], b1[2], b3[2];
#pragma unroll
        for (int i = 0; i < 4; ++i) {
            const int row = wm + (i << 4) + lm;
            a[i].q[0] = *(const uint4*)&As[cur][row][lh << 3];
            a[i].q[1] = *(const uint4*)&As[cur][row][16 + (lh << 3)];
        }
#pragma unroll
        for (int j = 0; j < 2; ++j) {
            const int col = wn + (j << 4) + lm;
            b1[j].q[0] = *(const uint4*)&B1s[cur][col][lh << 3];
            b1[j].q[1] = *(const uint4*)&B1s[cur][col][16 + (lh << 3)];
            b3[j].q[0] = *(const uint4*)&B3s[cur][col][lh << 3];
            b3[j].q[1] = *(const uint4*)&B3s[cur][col][16 + (lh << 3)];
        }
#pragma unroll
        for (int i = 0; i < 4; ++i)
#pragma unroll
            for (int j = 0; j < 2; ++j) {
                acc1[i][j] = wmma_bf16(a[i], b1[j], acc1[i][j]);
                acc3[i][j] = wmma_bf16(a[i], b3[j], acc3[i][j]);
            }
        __syncthreads();
    }
#pragma unroll
    for (int i = 0; i < 4; ++i)
#pragma unroll
        for (int j = 0; j < 2; ++j)
#pragma unroll
            for (int r = 0; r < 8; ++r) {
                const int row = bm + wm + (i << 4) + (lh << 3) + r;
                const int col = bn + wn + (j << 4) + lm;
                const float x1 = acc1[i][j][r];
                const float x3 = acc3[i][j][r];
                // silu via v_rcp_f32 (avoid IEEE division chains)
                const float sig = __builtin_amdgcn_rcpf(1.f + __expf(-x1));
                G[(size_t)row * N + col] = f2bf(x1 * sig * x3);
            }
}

// ---------------------------------------------------------------------------
// Flash attention: one wave per (head, 16-query tile). Causal, online softmax.
// qb already scaled by 1/sqrt(HDIM). Outputs bf16 [s][head*128+d].
// ---------------------------------------------------------------------------
__global__ __launch_bounds__(32) void attn_kernel(
    const u16* __restrict__ qb, const u16* __restrict__ kb,
    const u16* __restrict__ vt, u16* __restrict__ aob) {
    __shared__ __align__(16) u16 Ps[16][16];
    const int lane = threadIdx.x;
    const int lm = lane & 15, lh = lane >> 4;
    const int qt = blockIdx.x;          // query tile (16 rows)
    const int head = blockIdx.y;
    const int q0 = qt << 4;
    const int hbase = head << 7;

    // Q fragments for K-dim = 128 (4 chunks of 32)
    FragBF qfr[4];
#pragma unroll
    for (int c = 0; c < 4; ++c) {
        const u16* qp = qb + (size_t)(q0 + lm) * DIM + hbase + (c << 5) + (lh << 3);
        qfr[c].q[0] = *(const uint4*)qp;
        qfr[c].q[1] = *(const uint4*)(qp + 16);
    }

    v8f o[8] = {};
    float Mrow[8], Lrow[8];
#pragma unroll
    for (int r = 0; r < 8; ++r) { Mrow[r] = -INFINITY; Lrow[r] = 0.f; }

    for (int kt = 0; kt <= qt; ++kt) {
        // prefetch next key tile rows (global_prefetch_b8)
        if (kt + 1 <= qt)
            __builtin_prefetch(kb + (size_t)(((kt + 1) << 4) + lm) * DIM + hbase, 0, 3);

        FragBF kfr[4];
#pragma unroll
        for (int c = 0; c < 4; ++c) {
            const u16* kp = kb + (size_t)((kt << 4) + lm) * DIM + hbase + (c << 5) + (lh << 3);
            kfr[c].q[0] = *(const uint4*)kp;
            kfr[c].q[1] = *(const uint4*)(kp + 16);
        }
        v8f s = {};
#pragma unroll
        for (int c = 0; c < 4; ++c) s = wmma_bf16(qfr[c], kfr[c], s);

        if (kt == qt) {                      // causal mask on diagonal tile
#pragma unroll
            for (int r = 0; r < 8; ++r)
                if (lm > ((lh << 3) + r)) s[r] = -INFINITY;
        }

        float alpha[8];
#pragma unroll
        for (int r = 0; r < 8; ++r) {
            float mx = s[r];
#pragma unroll
            for (int msk = 1; msk < 16; msk <<= 1)
                mx = fmaxf(mx, __shfl_xor(mx, msk, 32));
            const float Mn = fmaxf(Mrow[r], mx);
            const float a = __expf(Mrow[r] - Mn);
            float pv = __expf(s[r] - Mn);
            s[r] = pv;
#pragma unroll
            for (int msk = 1; msk < 16; msk <<= 1)
                pv += __shfl_xor(pv, msk, 32);
            Lrow[r] = Lrow[r] * a + pv;
            Mrow[r] = Mn;
            alpha[r] = a;
        }
#pragma unroll
        for (int j = 0; j < 8; ++j)
#pragma unroll
            for (int r = 0; r < 8; ++r) o[j][r] *= alpha[r];

        // transpose P (C-layout -> A-layout) through LDS
        __syncthreads();
#pragma unroll
        for (int r = 0; r < 8; ++r) Ps[(lh << 3) + r][lm] = f2bf(s[r]);
        __syncthreads();

        FragBF pf;
        pf.q[0] = *(const uint4*)&Ps[lm][lh << 3];
        pf.q[1] = make_uint4(0, 0, 0, 0);           // K=16..31 zero pad
#pragma unroll
        for (int j = 0; j < 8; ++j) {
            FragBF vfr;
            const u16* vp = vt + (size_t)(hbase + (j << 4) + lm) * SEQ + (kt << 4) + (lh << 3);
            vfr.q[0] = *(const uint4*)vp;
            vfr.q[1] = make_uint4(0, 0, 0, 0);
            o[j] = wmma_bf16(pf, vfr, o[j]);
        }
    }
    float rl[8];
#pragma unroll
    for (int r = 0; r < 8; ++r) rl[r] = __builtin_amdgcn_rcpf(Lrow[r]);
#pragma unroll
    for (int j = 0; j < 8; ++j)
#pragma unroll
        for (int r = 0; r < 8; ++r) {
            const int row = q0 + (lh << 3) + r;
            const int col = hbase + (j << 4) + lm;
            aob[(size_t)row * DIM + col] = f2bf(o[j][r] * rl[r]);
        }
}

// ---------------------------------------------------------------------------
// Orchestration
// ---------------------------------------------------------------------------
extern "C" void kernel_launch(void* const* d_in, const int* in_sizes, int n_in,
                              void* d_out, int out_size, void* d_ws, size_t ws_size,
                              hipStream_t stream) {
    (void)in_sizes; (void)n_in; (void)out_size; (void)ws_size;
    const float* x   = (const float*)d_in[0];
    const float* fc  = (const float*)d_in[1];
    const float* fs  = (const float*)d_in[2];
    const float* nw1 = (const float*)d_in[3];
    const float* wq  = (const float*)d_in[4];
    const float* wk  = (const float*)d_in[5];
    const float* wv  = (const float*)d_in[6];
    const float* wo  = (const float*)d_in[7];
    const float* nw2 = (const float*)d_in[8];
    const float* w1  = (const float*)d_in[9];
    const float* w2  = (const float*)d_in[10];
    const float* w3  = (const float*)d_in[11];
    float* out = (float*)d_out;

    char* ws = (char*)d_ws;
    size_t off = 0;
    auto alloc = [&](size_t bytes) -> void* {
        void* p = ws + off;
        off = (off + bytes + 255) & ~(size_t)255;
        return p;
    };
    const size_t D2 = (size_t)DIM * DIM;     // 4M
    const size_t DH = (size_t)DIM * HID;     // 16M

    u16* wqT  = (u16*)alloc(D2 * 2);
    u16* wkT  = (u16*)alloc(D2 * 2);
    u16* wvT  = (u16*)alloc(D2 * 2);
    u16* woT  = (u16*)alloc(D2 * 2);
    u16* w1T  = (u16*)alloc(DH * 2);
    u16* w3T  = (u16*)alloc(DH * 2);
    u16* w2T  = (u16*)alloc(DH * 2);
    u16* xnb  = (u16*)alloc(D2 * 2);
    float* qf = (float*)alloc(D2 * 4);
    float* kf = (float*)alloc(D2 * 4);
    float* vf = (float*)alloc(D2 * 4);
    u16* qb   = (u16*)alloc(D2 * 2);
    u16* kb   = (u16*)alloc(D2 * 2);
    u16* vtb  = (u16*)alloc(D2 * 2);
    u16* aob  = (u16*)alloc(D2 * 2);
    float* h  = (float*)alloc(D2 * 4);
    u16* hnb  = (u16*)alloc(D2 * 2);
    u16* g    = (u16*)alloc(DH * 2);

    const dim3 tb(32, 8);
    // weight transpose+convert (grid = (N/32, K/32))
    transpose_cvt_kernel<<<dim3(DIM / 32, DIM / 32), tb, 0, stream>>>(wq, wqT, DIM, DIM);
    transpose_cvt_kernel<<<dim3(DIM / 32, DIM / 32), tb, 0, stream>>>(wk, wkT, DIM, DIM);
    transpose_cvt_kernel<<<dim3(DIM / 32, DIM / 32), tb, 0, stream>>>(wv, wvT, DIM, DIM);
    transpose_cvt_kernel<<<dim3(DIM / 32, DIM / 32), tb, 0, stream>>>(wo, woT, DIM, DIM);
    transpose_cvt_kernel<<<dim3(HID / 32, DIM / 32), tb, 0, stream>>>(w1, w1T, DIM, HID);
    transpose_cvt_kernel<<<dim3(HID / 32, DIM / 32), tb, 0, stream>>>(w3, w3T, DIM, HID);
    transpose_cvt_kernel<<<dim3(DIM / 32, HID / 32), tb, 0, stream>>>(w2, w2T, HID, DIM);

    // attn pre-norm
    rmsnorm_kernel<<<SEQ, 256, 0, stream>>>(x, nw1, xnb);

    // QKV projections
    gemm_bf16_f32<false><<<dim3(DIM / BN, SEQ / BM), 256, 0, stream>>>(xnb, wqT, nullptr, qf, SEQ, DIM, DIM);
    gemm_bf16_f32<false><<<dim3(DIM / BN, SEQ / BM), 256, 0, stream>>>(xnb, wkT, nullptr, kf, SEQ, DIM, DIM);
    gemm_bf16_f32<false><<<dim3(DIM / BN, SEQ / BM), 256, 0, stream>>>(xnb, wvT, nullptr, vf, SEQ, DIM, DIM);

    // RoPE (+1/sqrt(HDIM) folded into q), V transpose
    const int ropeBlocks = (SEQ * DIM / 2) / 256;
    rope_cvt_kernel<<<ropeBlocks, 256, 0, stream>>>(qf, fc, fs, qb, 0.08838834764831845f);
    rope_cvt_kernel<<<ropeBlocks, 256, 0, stream>>>(kf, fc, fs, kb, 1.0f);
    vtrans_kernel<<<(SEQ * DIM) / 256, 256, 0, stream>>>(vf, vtb);

    // flash attention
    attn_kernel<<<dim3(SEQ / 16, NHEADS), 32, 0, stream>>>(qb, kb, vtb, aob);

    // output projection + residual
    gemm_bf16_f32<true><<<dim3(DIM / BN, SEQ / BM), 256, 0, stream>>>(aob, woT, x, h, SEQ, DIM, DIM);

    // ffn pre-norm
    rmsnorm_kernel<<<SEQ, 256, 0, stream>>>(h, nw2, hnb);

    // fused SwiGLU (w1 & w3)
    gemm_swiglu<<<dim3(HID / BN, SEQ / BM), 256, 0, stream>>>(hnb, w1T, w3T, g, SEQ, HID, DIM);

    // down projection + residual -> out
    gemm_bf16_f32<true><<<dim3(DIM / BN, SEQ / BM), 256, 0, stream>>>(g, w2T, h, out, SEQ, DIM, HID);
}
